// Encoder_57320633532535
// MI455X (gfx1250) — matrix operand order repaired and checked by reading, not compiled, not run on hardware
//
#include <hip/hip_runtime.h>
#include <hip/hip_bf16.h>
#include <math.h>

// ---------------------------------------------------------------------------
// MI455X (gfx1250) implementation. wave32, WMMA 16x16x32 bf16 (f32 accum).
//
// B=64, EL=128, N=256, H=128, R=12, NB=8.
// - RGCN aggregation exploits closed-form sparsity (<=4 edges/node): fused
//   gather into LDS message tiles + 16x128 @ 128x128 WMMA (~1.6 GF effective
//   vs 26 GF dense).
// - Weights pre-packed FRAGMENT-MAJOR: each (colblock,k4,lane) owns 8
//   contiguous uints -> B-fragment = one 32B load (2x global_load_b128).
// - Dominant cost is the 218MB int32 adjacency output (> 192MB L2):
//   non-temporal 128-bit stores.
// ---------------------------------------------------------------------------

#define BSZ   64
#define ELEN  128
#define NNODE 256
#define HDIM  128
#define NREL  12
#define NBAS  8
#define LSEQ  127   // EL - 1
#define WMAT  8192  // uints per packed 128x128 weight matrix

typedef __attribute__((ext_vector_type(16))) __bf16 v16bf;
typedef __attribute__((ext_vector_type(8)))  float  v8f;
typedef __attribute__((ext_vector_type(4)))  int    vi4;
typedef __attribute__((ext_vector_type(8)))  unsigned int vu8;

// ---------------- WMMA fragment helpers (contiguous K order, same in A and B) -----

// A-matrix 16x32 bf16: lane<16 -> M=lane, V0..7:K=0..15;
// lanes 16..31 add +16 to K. Contiguous K pairs -> compiler emits ds_load_b128.
__device__ __forceinline__ v16bf load_a_frag(const __bf16* tile /*16 x HDIM*/,
                                             int mrow, int hf, int kbase) {
  v16bf a;
  const __bf16* row = tile + mrow * HDIM;
#pragma unroll
  for (int v = 0; v < 8; ++v) {
    int K = kbase + 2 * v + hf * 16;
    a[2 * v]     = row[K];
    a[2 * v + 1] = row[K + 1];
  }
  return a;
}

// B-matrix 32x16 bf16, fragment-major packed: lane's 8 uints are contiguous.
// uint index = ((wave*4 + k4)*32 + lane)*8 + v, where each uint packs
// (K,K+1) for col = wave*16+(lane&15), K = 32*k4 + (lane>>4)*16 + 2v.
__device__ __forceinline__ v16bf load_b_frag(const unsigned* wfrag, int wave,
                                             int lane, int k4) {
  const vu8* p = (const vu8*)(wfrag + ((((wave << 2) + k4) << 5) + lane) * 8);
  return __builtin_bit_cast(v16bf, *p);
}

__device__ __forceinline__ unsigned pack_bf16_pair(float f0, float f1) {
  unsigned short u0 = __builtin_bit_cast(unsigned short, (__bf16)f0);
  unsigned short u1 = __builtin_bit_cast(unsigned short, (__bf16)f1);
  return (unsigned)u0 | ((unsigned)u1 << 16);
}

// ---------------- edge enumeration (closed form, <=4 edges per node) --------

struct Edge { int r; int m; };

// relations r = channel-1 (channel 0 dropped). structural: n<->n+2 (ch 11/12),
// dynamic: n%4==0 -> j=n+3 (s=n/2), as-j i=n-1 (s=n/2-1);
//          n%4==3 -> j=n+1 (s=(n-1)/2), as-j i=n-3 (s=(n-3)/2).
__device__ __forceinline__ int get_edges(int n, const int* st, Edge* e) {
  int cnt = 0;
  int ni = n >> 1;
  if (n + 2 < NNODE) {
    int ch = ((n & 1) == 0) ? (((ni & 1) == 0) ? 11 : 12)
                            : (((ni & 1) == 0) ? 12 : 11);
    e[cnt].r = ch - 1; e[cnt].m = n + 2; ++cnt;
  }
  if (n - 2 >= 0) {
    int nim = ni - 1;
    int ch = ((n & 1) == 0) ? (((nim & 1) == 0) ? 11 : 12)
                            : (((nim & 1) == 0) ? 12 : 11);
    e[cnt].r = ch - 1; e[cnt].m = n - 2; ++cnt;
  }
  int mod = n & 3;
  if (mod == 0) {
    int s = n >> 1;
    if (s < LSEQ) { int ch = st[s]; if (ch >= 1) { e[cnt].r = ch - 1; e[cnt].m = n + 3; ++cnt; } }
    int s2 = (n >> 1) - 1;
    if (s2 >= 0 && s2 < LSEQ) { int ch = st[s2]; if (ch >= 1) { e[cnt].r = ch - 1; e[cnt].m = n - 1; ++cnt; } }
  } else if (mod == 3) {
    int s = (n - 1) >> 1;
    if (s < LSEQ) { int ch = st[s]; if (ch >= 1) { e[cnt].r = ch - 1; e[cnt].m = n + 1; ++cnt; } }
    int s2 = (n - 3) >> 1;
    if (s2 >= 0 && s2 < LSEQ) { int ch = st[s2]; if (ch >= 1) { e[cnt].r = ch - 1; e[cnt].m = n - 3; ++cnt; } }
  }
  return cnt;
}

// ---------------- weight prep: fragment-major bf16 packing ------------------

// For x @ W.T style GEMMs: B[K][col] = in[col][K].
__global__ void pack_T_kernel(const float* __restrict__ in,
                              unsigned* __restrict__ out) {
  int idx = blockIdx.x * blockDim.x + threadIdx.x;
  if (idx >= WMAT) return;
  int v = idx & 7, lane = (idx >> 3) & 31, k4 = (idx >> 8) & 3, w = idx >> 10;
  int col = w * 16 + (lane & 15);
  int K   = 32 * k4 + (lane >> 4) * 16 + 2 * v;
  out[idx] = pack_bf16_pair(in[col * HDIM + K], in[col * HDIM + K + 1]);
}

// Wrel[r,K,col] = sum_nb comb[r,nb] * basis[nb,K,col]  (B[K][col] directly).
__global__ void pack_wrel_kernel(const float* __restrict__ basis,
                                 const float* __restrict__ comb,
                                 unsigned* __restrict__ out) {
  int idx = blockIdx.x * blockDim.x + threadIdx.x;
  if (idx >= NREL * WMAT) return;
  int t = idx & (WMAT - 1);
  int r = idx >> 13;
  int v = t & 7, lane = (t >> 3) & 31, k4 = (t >> 8) & 3, w = t >> 10;
  int col = w * 16 + (lane & 15);
  int K   = 32 * k4 + (lane >> 4) * 16 + 2 * v;
  float f0 = 0.f, f1 = 0.f;
#pragma unroll
  for (int nb = 0; nb < NBAS; ++nb) {
    float c = comb[r * NBAS + nb];
    f0 += c * basis[((size_t)nb * HDIM + K) * HDIM + col];
    f1 += c * basis[((size_t)nb * HDIM + K + 1) * HDIM + col];
  }
  out[idx] = pack_bf16_pair(f0, f1);
}

// ---------------- input projection: rally @ in_W.T + in_b  ->  x0 (bf16) ----

__global__ __launch_bounds__(256) void input_kernel(
    const int* __restrict__ player, const float* __restrict__ Ax,
    const float* __restrict__ Ay, const float* __restrict__ Bx,
    const float* __restrict__ By, const float* __restrict__ emb,
    const float* __restrict__ coordW, const float* __restrict__ coordb,
    const unsigned* __restrict__ inWP, const float* __restrict__ inb,
    __bf16* __restrict__ x0) {
  __shared__ __bf16 rally[16][HDIM];

  int tid = threadIdx.x;
  int b  = blockIdx.x >> 4;
  int n0 = (blockIdx.x & 15) * 16;
  int i  = tid >> 4;
  int c0 = (tid & 15) * 8;
  int n  = n0 + i;
  int t  = n >> 1;

  float cx, cy;
  if ((n & 1) == 0) { cx = Ax[b * ELEN + t]; cy = Ay[b * ELEN + t]; }
  else              { cx = Bx[b * ELEN + t]; cy = By[b * ELEN + t]; }

  if (c0 < 64) {  // ct = relu(coords @ coord_W.T + coord_b)
    for (int c = c0; c < c0 + 8; ++c) {
      float v = coordW[c * 2 + 0] * cx + coordW[c * 2 + 1] * cy + coordb[c];
      rally[i][c] = (__bf16)fmaxf(v, 0.f);
    }
  } else {        // pe = emb_table[player[b, n&1]]
    int pid = player[b * 2 + (n & 1)];
    const float* er = emb + (size_t)pid * 64;
    for (int c = c0; c < c0 + 8; ++c) rally[i][c] = (__bf16)er[c - 64];
  }
  __syncthreads();

  int wave = tid >> 5, lane = tid & 31;
  int hf = lane >> 4, mrow = lane & 15;
  int colg = wave * 16 + (lane & 15);

  v8f acc = {};
#pragma unroll
  for (int k4 = 0; k4 < 4; ++k4) {
    v16bf a = load_a_frag(&rally[0][0], mrow, hf, k4 * 32);
    v16bf w = load_b_frag(inWP, wave, lane, k4);
    acc = __builtin_amdgcn_wmma_f32_16x16x32_bf16(false, a, false, w,
                                                  (short)0, acc, false, false);
  }
  float bias = inb[colg];
#pragma unroll
  for (int v = 0; v < 8; ++v) {
    int row = hf * 8 + v;
    size_t o = ((size_t)b * NNODE + (n0 + row)) * HDIM + colg;
    x0[o] = (__bf16)(acc[v] + bias);
  }
}

// ---------------- fused RGCN layer ------------------------------------------

__global__ __launch_bounds__(256) void layer_kernel(
    const __bf16* __restrict__ x_in,      // [B,N,H] bf16
    const unsigned* __restrict__ wrelP,   // [12] fragment-major 128x128
    const unsigned* __restrict__ wselfP,  // fragment-major 128x128 (= self.T)
    const int* __restrict__ shot_type,    // [B,127]
    float* __restrict__ out_f32,          // may be null
    __bf16* __restrict__ out_bf16,        // may be null
    int act) {                             // 0 = relu, 1 = sigmoid
  __shared__ __bf16 msg[13][16][HDIM];     // slots 0..11 = relations, 12 = self
  __shared__ unsigned relmask_sh;

  int tid = threadIdx.x;
  int b  = blockIdx.x >> 4;
  int n0 = (blockIdx.x & 15) * 16;
  if (tid == 0) relmask_sh = 0u;
  __syncthreads();

  // ---- phase 1: sparse gather into per-relation message tiles --------------
  int i  = tid >> 4;
  int c0 = (tid & 15) * 8;
  int n  = n0 + i;
  const int* st = shot_type + b * LSEQ;

  Edge e[4];
  int ne = get_edges(n, st, e);
  unsigned m = 0;
  for (int k = 0; k < ne; ++k) m |= 1u << e[k].r;
  atomicOr(&relmask_sh, m);

  const __bf16* xrow = x_in + ((size_t)b * NNODE + n) * HDIM;
  for (int c = c0; c < c0 + 8; ++c) msg[12][i][c] = xrow[c];
#pragma unroll
  for (int r = 0; r < NREL; ++r)
    for (int c = c0; c < c0 + 8; ++c) msg[r][i][c] = (__bf16)0.0f;
  for (int k = 0; k < ne; ++k) {
    const __bf16* xm = x_in + ((size_t)b * NNODE + e[k].m) * HDIM;
    int r = e[k].r;
    for (int c = c0; c < c0 + 8; ++c)
      msg[r][i][c] = (__bf16)((float)msg[r][i][c] + (float)xm[c]);
  }
  __syncthreads();
  unsigned relmask = relmask_sh;

  // ---- phase 2: WMMA (self + each present relation) ------------------------
  int wave = tid >> 5, lane = tid & 31;
  int hf = lane >> 4, mrow = lane & 15;
  int colg = wave * 16 + (lane & 15);

  v8f acc = {};
#pragma unroll
  for (int k4 = 0; k4 < 4; ++k4) {  // x @ self.T
    v16bf a = load_a_frag(&msg[12][0][0], mrow, hf, k4 * 32);
    v16bf w = load_b_frag(wselfP, wave, lane, k4);
    acc = __builtin_amdgcn_wmma_f32_16x16x32_bf16(false, a, false, w,
                                                  (short)0, acc, false, false);
  }
  for (int r = 0; r < NREL; ++r) {
    if (!(relmask & (1u << r))) continue;
    const unsigned* w = wrelP + (size_t)r * WMAT;
#pragma unroll
    for (int k4 = 0; k4 < 4; ++k4) {
      v16bf a  = load_a_frag(&msg[r][0][0], mrow, hf, k4 * 32);
      v16bf wf = load_b_frag(w, wave, lane, k4);
      acc = __builtin_amdgcn_wmma_f32_16x16x32_bf16(false, a, false, wf,
                                                    (short)0, acc, false, false);
    }
  }

  // ---- phase 3: activation + store -----------------------------------------
#pragma unroll
  for (int v = 0; v < 8; ++v) {
    float val = acc[v];
    val = (act == 0) ? fmaxf(val, 0.f) : 1.f / (1.f + expf(-val));
    int row = hf * 8 + v;
    size_t o = ((size_t)b * NNODE + (n0 + row)) * HDIM + colg;
    if (out_f32)  out_f32[o]  = val;
    if (out_bf16) out_bf16[o] = (__bf16)val;
  }
}

// ---------------- adjacency output (int32, NT stores) -----------------------

__global__ void adj_zero_kernel(vi4* __restrict__ p, long n4) {
  long idx = (long)blockIdx.x * blockDim.x + threadIdx.x;
  long stride = (long)gridDim.x * blockDim.x;
  vi4 z = {0, 0, 0, 0};
  for (; idx < n4; idx += stride) __builtin_nontemporal_store(z, p + idx);
}

__global__ void adj_set_kernel(int* __restrict__ adj, const int* __restrict__ shot) {
  int tid = blockIdx.x * blockDim.x + threadIdx.x;
  int b = tid / (NNODE + LSEQ);
  int k = tid % (NNODE + LSEQ);
  if (b >= BSZ) return;
  size_t bb = (size_t)b * 13 * NNODE * NNODE;
  if (k < NNODE) {  // structural edge n <-> n+2
    int nn = k;
    if (nn + 2 < NNODE) {
      int ni = nn >> 1;
      int ch = ((nn & 1) == 0) ? (((ni & 1) == 0) ? 11 : 12)
                               : (((ni & 1) == 0) ? 12 : 11);
      size_t base = bb + (size_t)ch * NNODE * NNODE;
      adj[base + (size_t)nn * NNODE + nn + 2] = 1;
      adj[base + (size_t)(nn + 2) * NNODE + nn] = 1;
    }
  } else {          // dynamic edge from shot_type
    int s = k - NNODE;
    int ch = shot[b * LSEQ + s];
    int ii, jj;
    if ((s & 1) == 0) { ii = 2 * s;     jj = 2 * s + 3; }
    else              { ii = 2 * s + 1; jj = 2 * s + 2; }
    size_t base = bb + (size_t)ch * NNODE * NNODE;
    adj[base + (size_t)ii * NNODE + jj] = 1;
    adj[base + (size_t)jj * NNODE + ii] = 1;
  }
}

// ---------------------------------------------------------------------------

extern "C" void kernel_launch(void* const* d_in, const int* in_sizes, int n_in,
                              void* d_out, int out_size, void* d_ws, size_t ws_size,
                              hipStream_t stream) {
  (void)in_sizes; (void)n_in; (void)out_size; (void)ws_size;

  const int*   player = (const int*)d_in[0];
  const int*   shot   = (const int*)d_in[1];
  const float* Ax     = (const float*)d_in[2];
  const float* Ay     = (const float*)d_in[3];
  const float* Bx     = (const float*)d_in[4];
  const float* By     = (const float*)d_in[5];
  //            d_in[6] = encode_length (128, compile-time here)
  const float* emb    = (const float*)d_in[7];
  const float* coordW = (const float*)d_in[8];
  const float* coordb = (const float*)d_in[9];
  const float* inW    = (const float*)d_in[10];
  const float* inb    = (const float*)d_in[11];
  const float* basis0 = (const float*)d_in[12];
  const float* comb0  = (const float*)d_in[13];
  const float* self0  = (const float*)d_in[14];
  const float* basis1 = (const float*)d_in[15];
  const float* comb1  = (const float*)d_in[16];
  const float* self1  = (const float*)d_in[17];

  // workspace carve-up (~9.2 MB)
  char* ws = (char*)d_ws;
  unsigned* wrelP0  = (unsigned*)ws; ws += (size_t)NREL * WMAT * 4;
  unsigned* wselfP0 = (unsigned*)ws; ws += (size_t)WMAT * 4;
  unsigned* wrelP1  = (unsigned*)ws; ws += (size_t)NREL * WMAT * 4;
  unsigned* wselfP1 = (unsigned*)ws; ws += (size_t)WMAT * 4;
  unsigned* inWP    = (unsigned*)ws; ws += (size_t)WMAT * 4;
  __bf16*   x0      = (__bf16*)ws;   ws += (size_t)BSZ * NNODE * HDIM * 2;
  __bf16*   x1      = (__bf16*)ws;   ws += (size_t)BSZ * NNODE * HDIM * 2;

  // 1) weight prep / packing
  pack_T_kernel<<<(WMAT + 255) / 256, 256, 0, stream>>>(inW, inWP);
  pack_T_kernel<<<(WMAT + 255) / 256, 256, 0, stream>>>(self0, wselfP0);
  pack_T_kernel<<<(WMAT + 255) / 256, 256, 0, stream>>>(self1, wselfP1);
  pack_wrel_kernel<<<(NREL * WMAT + 255) / 256, 256, 0, stream>>>(basis0, comb0, wrelP0);
  pack_wrel_kernel<<<(NREL * WMAT + 255) / 256, 256, 0, stream>>>(basis1, comb1, wrelP1);

  // 2) input projection (WMMA)
  input_kernel<<<BSZ * (NNODE / 16), 256, 0, stream>>>(
      player, Ax, Ay, Bx, By, emb, coordW, coordb, inWP, inb, x0);

  // 3) two fused RGCN layers (WMMA)
  layer_kernel<<<BSZ * (NNODE / 16), 256, 0, stream>>>(
      x0, wrelP0, wselfP0, shot, (float*)nullptr, x1, /*relu*/0);
  layer_kernel<<<BSZ * (NNODE / 16), 256, 0, stream>>>(
      x1, wrelP1, wselfP1, shot, (float*)d_out, (__bf16*)nullptr, /*sigmoid*/1);

  // 4) adjacency output: int32 region after x ([B,N,H] f32)
  int* adj = (int*)((float*)d_out + (size_t)BSZ * NNODE * HDIM);
  long n_adj = (long)BSZ * 13 * NNODE * NNODE;   // 54,525,952
  adj_zero_kernel<<<4096, 256, 0, stream>>>((vi4*)adj, n_adj / 4);
  adj_set_kernel<<<(BSZ * (NNODE + LSEQ) + 255) / 256, 256, 0, stream>>>(adj, shot);
}
